// SemanticBatchHardXBM2_33268816675424
// MI455X (gfx1250) — compile-verified, matrix operand order checked by reading, not run on hardware
//
#include <hip/hip_runtime.h>
#include <math.h>

// ---------------------------------------------------------------------------
// Problem constants (from reference): B=256, WV=300, D=512, K=65536, 32 classes
// d_in: 0:x[B,D] 1:wv[B,WV] 2:label[B] 3:alpha[B,1] 4:W[D,WV] 5:b[D]
//       6:x_xbm[K,D] 7:label_xbm[K] 8:it (int scalar)
// d_out: diff[B] f32  ++  sem[B,D] f32
// ---------------------------------------------------------------------------
#define B_  256
#define WV_ 300
#define D_  512
#define K_  65536
#define NBLK 64       // xbm rows per workgroup (4 n-tiles of 16)
#define KCHUNK 32     // K per WMMA step
#define NKIT (D_ / KCHUNK)
#define LDA 40        // padded LDS row stride in bf16 elems (32 + 8)

typedef __attribute__((ext_vector_type(16))) __bf16 v16bf;
typedef __attribute__((ext_vector_type(8)))  float  v8f;

union FragU { v16bf v; uint4 q[2]; };

// --------------------------- init fp / cn bits ------------------------------
__global__ void k_init(unsigned* fpBits, unsigned* cnBits) {
    int t = threadIdx.x;
    fpBits[t] = 0u;                 // max identity for non-negative floats
    cnBits[t] = 0x7F7FFFFFu;        // +FLT_MAX bits, min identity
}

// --------- sem = alpha*(wv@W.T + b) + (1-alpha)*x ; bf16 copy + norms -------
__global__ void k_sem(const float* __restrict__ x, const float* __restrict__ wv,
                      const float* __restrict__ alpha, const float* __restrict__ W,
                      const float* __restrict__ b, float* __restrict__ sem_out,
                      __bf16* __restrict__ semb, float* __restrict__ semNorm) {
    __shared__ float swv[WV_];
    __shared__ float red[256];
    const int i = blockIdx.x, t = threadIdx.x;
    for (int k = t; k < WV_; k += 256) swv[k] = wv[i * WV_ + k];
    __syncthreads();
    const float a = alpha[i];
    float ss = 0.f;
    for (int half = 0; half < 2; ++half) {
        const int c = t + half * 256;
        const float* wr = &W[c * WV_];
        float dot = 0.f;
        for (int k = 0; k < WV_; ++k) dot = fmaf(swv[k], wr[k], dot);
        const float s = a * (dot + b[c]) + (1.f - a) * x[i * D_ + c];
        sem_out[i * D_ + c] = s;
        semb[i * D_ + c]    = (__bf16)s;
        ss = fmaf(s, s, ss);
    }
    red[t] = ss;
    __syncthreads();
    for (int off = 128; off > 0; off >>= 1) {
        if (t < off) red[t] += red[t + off];
        __syncthreads();
    }
    if (t == 0) semNorm[i] = red[0];
}

// ----------------- intra-batch 256x256 distances + mask reduce --------------
__global__ void k_intra(const float* __restrict__ sem, const float* __restrict__ x,
                        const int* __restrict__ label, const float* __restrict__ semNorm,
                        unsigned* fpBits, unsigned* cnBits) {
    __shared__ float ssem[D_];
    __shared__ float rmax[256];
    __shared__ float rmin[256];
    const int i = blockIdx.x, j = threadIdx.x;
    for (int k = j; k < D_; k += 256) ssem[k] = sem[i * D_ + k];
    __syncthreads();
    const float* xr = &x[j * D_];
    float dot = 0.f, xn = 0.f;
    for (int k = 0; k < D_; ++k) {
        const float xv = xr[k];
        dot = fmaf(ssem[k], xv, dot);
        xn  = fmaf(xv, xv, xn);
    }
    const float sq   = semNorm[i] + xn - 2.f * dot;
    const float dist = sqrtf(fmaxf(sq, 0.f) + (float)D_ * 1e-16f);
    const bool same  = (label[i] == label[j]);
    const bool pos   = same && (i != j);
    rmax[j] = pos ? dist : 0.f;
    rmin[j] = dist + (same ? 1e8f : 0.f);
    __syncthreads();
    for (int off = 128; off > 0; off >>= 1) {
        if (j < off) {
            rmax[j] = fmaxf(rmax[j], rmax[j + off]);
            rmin[j] = fminf(rmin[j], rmin[j + off]);
        }
        __syncthreads();
    }
    if (j == 0) {
        atomicMax(&fpBits[i], __float_as_uint(rmax[0]));
        atomicMin(&cnBits[i], __float_as_uint(rmin[0]));
    }
}

// ---- XBM block: 256 x 65536 distances via bf16 WMMA, double-buffered -------
// Register discipline: only the HBM-stream (B) staging regs live across the
// WMMA block; the L2-hot A chunk is loaded+stored to LDS immediately.
__global__ void __launch_bounds__(256, 1)
k_xbm(const __bf16* __restrict__ semb, const float* __restrict__ x_xbm,
      const int* __restrict__ label, const int* __restrict__ label_xbm,
      const float* __restrict__ semNorm, const int* __restrict__ itp,
      unsigned* fpBits, unsigned* cnBits) {
    if (*itp <= 1000) return;    // reference gate; uniform per launch

    __shared__ __align__(16) __bf16 lA[2][B_ * LDA];    // 2 x 20 KB
    __shared__ __align__(16) __bf16 lB[2][NBLK * LDA];  // 2 x 5 KB
    __shared__ float    normB[NBLK];                    // per-xbm-row |x|^2
    __shared__ unsigned fpL[B_];                        // block-local reduce
    __shared__ unsigned cnL[B_];

    const int t    = threadIdx.x;
    const int wave = t >> 5;
    const int lane = t & 31;
    const int N0   = blockIdx.x * NBLK;

    if (t < NBLK) normB[t] = 0.f;
    fpL[t] = 0u;
    cnL[t] = 0x7F7FFFFFu;

    // wave tiling: 16 m-tiles x 4 n-tiles = 64 tiles, 8 tiles/wave (4m x 2n)
    const int mgrp = wave >> 1;          // 0..3 -> m-tiles mgrp*4 + q
    const int ngrp = wave & 1;           // 0..1 -> n-tiles ngrp*2 + p

    v8f acc[4][2];
    { v8f z = {}; for (int q = 0; q < 4; ++q) for (int p = 0; p < 2; ++p) acc[q][p] = z; }

    float4 rb_[2];                       // B staging regs (live across WMMAs)
    float  nsq0 = 0.f, nsq1 = 0.f;       // per-thread |x_xbm| partial sums

    // A: 256 rows x 32 bf16 per chunk, 4 threads/row -> load & store at once
    auto stageA = [&](int kit, int buf) {
        const int k0 = kit * KCHUNK;
#pragma unroll
        for (int j = 0; j < 4; ++j) {
            const int g = t + 256 * j;
            const uint4 v = *(const uint4*)(semb + (g >> 2) * D_ + k0 + (g & 3) * 8);
            *(uint4*)(&lA[buf][(g >> 2) * LDA + (g & 3) * 8]) = v;
        }
    };
    auto loadB = [&](int kit) {          // B: 64 rows x 32 f32, 8 threads/row
        const int k0 = kit * KCHUNK;
#pragma unroll
        for (int j = 0; j < 2; ++j) {
            const int e = t + 256 * j;
            const float4 f = *(const float4*)(x_xbm +
                (size_t)(N0 + (e >> 3)) * D_ + k0 + (e & 7) * 4);
            rb_[j] = f;
            const float s = f.x * f.x + f.y * f.y + f.z * f.z + f.w * f.w;
            if (j == 0) nsq0 += s; else nsq1 += s;
        }
    };
    auto storeB = [&](int buf) {
#pragma unroll
        for (int j = 0; j < 2; ++j) {
            const int e = t + 256 * j;
            union { __bf16 h[4]; uint2 u; } pk;
            pk.h[0] = (__bf16)rb_[j].x; pk.h[1] = (__bf16)rb_[j].y;
            pk.h[2] = (__bf16)rb_[j].z; pk.h[3] = (__bf16)rb_[j].w;
            *(uint2*)(&lB[buf][(e >> 3) * LDA + (e & 7) * 4]) = pk.u;
        }
    };
    auto compute = [&](int buf) {
        // wave32 16-bit fragment layout: lanes 0-15 K{0..7,16..23},
        // lanes 16-31 K{8..15,24..31}; two 16B runs per lane.
        const int kbase = (lane < 16) ? 0 : 8;
        FragU fb[2];
#pragma unroll
        for (int p = 0; p < 2; ++p) {
            const int rb = (ngrp * 2 + p) * 16 + (lane & 15);
            fb[p].q[0] = *(const uint4*)(&lB[buf][rb * LDA + kbase]);
            fb[p].q[1] = *(const uint4*)(&lB[buf][rb * LDA + kbase + 16]);
        }
#pragma unroll
        for (int q = 0; q < 4; ++q) {
            const int ra = (mgrp * 4 + q) * 16 + (lane & 15);
            FragU fa;
            fa.q[0] = *(const uint4*)(&lA[buf][ra * LDA + kbase]);
            fa.q[1] = *(const uint4*)(&lA[buf][ra * LDA + kbase + 16]);
#pragma unroll
            for (int p = 0; p < 2; ++p)
                acc[q][p] = __builtin_amdgcn_wmma_f32_16x16x32_bf16(
                    false, fa.v, false, fb[p].v, (short)0, acc[q][p], false, false);
        }
    };

    // ---- software pipeline: 1 barrier per K-step; B loads overlap WMMAs ----
    stageA(0, 0);
    loadB(0);
    storeB(0);
    for (int kit = 0; kit < NKIT; ++kit) {
        const int cur = kit & 1;
        __syncthreads();                 // buf[cur] ready; buf[1-cur] readers done
        const bool more = (kit + 1 < NKIT);
        if (more) {
            stageA(kit + 1, 1 - cur);    // L2-hot: load+store now, regs transient
            if (kit + 2 < NKIT)          // prefetch A chunk k+2 (global_prefetch)
                __builtin_prefetch(semb + (t >> 2) * D_ + (kit + 2) * KCHUNK, 0, 0);
            loadB(kit + 1);              // HBM stream stays in regs under WMMAs
        }
        compute(cur);
        if (more) storeB(1 - cur);       // safe: barrier retired old readers
    }

    // ---- flush |x_xbm|^2 partials, then epilogue ----
    atomicAdd(&normB[t >> 3], nsq0);
    atomicAdd(&normB[(t + 256) >> 3], nsq1);
    __syncthreads();

    const int nn0 = ngrp * 32 + (lane & 15);     // p=0 column, p=1 is +16
    const float xn0 = normB[nn0];
    const float xn1 = normB[nn0 + 16];
    const int   ln0 = label_xbm[N0 + nn0];
    const int   ln1 = label_xbm[N0 + nn0 + 16];
    const int   hi8 = (lane >> 4) << 3;          // lanes>=16 hold M rows 8..15
#pragma unroll
    for (int q = 0; q < 4; ++q) {
#pragma unroll
        for (int r = 0; r < 8; ++r) {
            const int m  = (mgrp * 4 + q) * 16 + hi8 + r;
            const float sn = semNorm[m];
            const int   lm = label[m];
            const float d0 = sqrtf(fmaxf(sn + xn0 - 2.f * acc[q][0][r], 0.f)
                                   + (float)D_ * 1e-16f);
            const float d1 = sqrtf(fmaxf(sn + xn1 - 2.f * acc[q][1][r], 0.f)
                                   + (float)D_ * 1e-16f);
            const bool p0 = (lm == ln0), p1 = (lm == ln1);
            float fpv = fmaxf(p0 ? d0 : 0.f, p1 ? d1 : 0.f);
            float cnv = fminf(d0 + (p0 ? 1e8f : 0.f), d1 + (p1 ? 1e8f : 0.f));
#pragma unroll
            for (int mk = 1; mk < 16; mk <<= 1) {
                fpv = fmaxf(fpv, __shfl_xor(fpv, mk, 32));
                cnv = fminf(cnv, __shfl_xor(cnv, mk, 32));
            }
            if ((lane & 15) == 0) {              // lanes 0 and 16 write
                atomicMax(&fpL[m], __float_as_uint(fpv));   // ds_max_u32
                atomicMin(&cnL[m], __float_as_uint(cnv));   // ds_min_u32
            }
        }
    }
    __syncthreads();
    atomicMax(&fpBits[t], fpL[t]);               // 512 global atomics / block
    atomicMin(&cnBits[t], cnL[t]);
}

// ------------------------------- softplus -----------------------------------
__global__ void k_final(const unsigned* __restrict__ fpBits,
                        const unsigned* __restrict__ cnBits, float* __restrict__ out) {
    const int i = threadIdx.x;
    const float fp = __uint_as_float(fpBits[i]);
    const float cn = __uint_as_float(cnBits[i]);
    const float v  = fp - cn;
    out[i] = (v > 20.f) ? v : log1pf(expf(v));
}

// ---------------------------------------------------------------------------
extern "C" void kernel_launch(void* const* d_in, const int* in_sizes, int n_in,
                              void* d_out, int out_size, void* d_ws, size_t ws_size,
                              hipStream_t stream) {
    const float* x         = (const float*)d_in[0];
    const float* wv        = (const float*)d_in[1];
    const int*   label     = (const int*)  d_in[2];
    const float* alpha     = (const float*)d_in[3];
    const float* W         = (const float*)d_in[4];
    const float* b         = (const float*)d_in[5];
    const float* x_xbm     = (const float*)d_in[6];
    const int*   label_xbm = (const int*)  d_in[7];
    const int*   itp       = (const int*)  d_in[8];

    float* diff_out = (float*)d_out;          // [256]
    float* sem_out  = (float*)d_out + B_;     // [256*512]

    // workspace layout: bf16 sem copy | sem norms | fp bits | cn bits
    char* ws = (char*)d_ws;
    __bf16*   semb    = (__bf16*)ws;                          // 262144 B
    float*    semNorm = (float*)(ws + (size_t)B_ * D_ * 2);   // 1 KB
    unsigned* fpBits  = (unsigned*)((char*)semNorm + B_ * sizeof(float));
    unsigned* cnBits  = fpBits + B_;

    k_init  <<<1,          256, 0, stream>>>(fpBits, cnBits);
    k_sem   <<<B_,         256, 0, stream>>>(x, wv, alpha, W, b,
                                             sem_out, semb, semNorm);
    k_intra <<<B_,         256, 0, stream>>>(sem_out, x, label, semNorm,
                                             fpBits, cnBits);
    k_xbm   <<<K_ / NBLK,  256, 0, stream>>>(semb, x_xbm, label, label_xbm,
                                             semNorm, itp, fpBits, cnBits);
    k_final <<<1,          256, 0, stream>>>(fpBits, cnBits, diff_out);
}
